// ResGCN_69965017252460
// MI455X (gfx1250) — matrix operand order repaired and checked by reading, not compiled, not run on hardware
//
#include <hip/hip_runtime.h>
#include <hip/hip_bf16.h>
#include <stdint.h>

typedef __attribute__((ext_vector_type(16))) _Float16 v16h;
typedef __attribute__((ext_vector_type(8)))  float    v8f;
typedef __attribute__((ext_vector_type(4)))  float    v4f;

#define NNODES 100000
#define NEDGES 1600000
#define NF     64
#define NGRAPH 512
#define NLAYERS 4

// ---------------------------------------------------------------- utilities
__global__ void zero_f32(float* __restrict__ p, int n) {
  int i = blockIdx.x * 256 + threadIdx.x;
  if (i < n) p[i] = 0.0f;
}

__global__ void deg_count(const int* __restrict__ dst, int* __restrict__ degi) {
  int e = blockIdx.x * 256 + threadIdx.x;
  if (e < NEDGES) atomicAdd(&degi[dst[e]], 1);
}

__global__ void deg_inv(const int* __restrict__ degi, float* __restrict__ dinv) {
  int i = blockIdx.x * 256 + threadIdx.x;
  if (i < NNODES) dinv[i] = rsqrtf((float)(degi[i] + 1));   // +1 self-loop
}

// norm[e] = dinv[src]*dinv[dst], computed once, reused by all 4 layers
__global__ void edge_norm(const int* __restrict__ src, const int* __restrict__ dst,
                          const float* __restrict__ dinv, float* __restrict__ norm) {
  int e = blockIdx.x * 256 + threadIdx.x;
  if (e < NEDGES) norm[e] = dinv[src[e]] * dinv[dst[e]];
}

// --------------------------------------------------- column sum / sumsq (big)
__global__ __launch_bounds__(256) void stats_accum(const float* __restrict__ X,
                                                   int rows,
                                                   float* __restrict__ accum) {
  __shared__ float rs[4][64], rq[4][64];
  const int col = threadIdx.x & 63;
  const int rg  = threadIdx.x >> 6;
  float s = 0.f, q = 0.f;
  for (int r = blockIdx.x * 4 + rg; r < rows; r += gridDim.x * 4) {
    float v = X[(size_t)r * NF + col];
    s += v; q += v * v;
  }
  rs[rg][col] = s; rq[rg][col] = q;
  __syncthreads();
  if (rg == 0) {
    s = rs[0][col] + rs[1][col] + rs[2][col] + rs[3][col];
    q = rq[0][col] + rq[1][col] + rq[2][col] + rq[3][col];
    atomicAdd(&accum[col], s);
    atomicAdd(&accum[64 + col], q);
  }
}

// ---- fold BN into weights: Wt[n][k] = f16(scale[k]*W[k][n]),
// ---- cinit[n] = (bias?bias[n]:0) + sum_k shift[k]*W[k][n]
__global__ __launch_bounds__(256) void fold_params(
    const float* __restrict__ accum,   // [128] sum/sumsq, or null
    const float* __restrict__ X,       // if non-null: compute stats here (small)
    int rows,
    const float* __restrict__ g, const float* __restrict__ b,
    const float* __restrict__ W,       // [64,64] k-major
    const float* __restrict__ bias,    // may be null
    _Float16* __restrict__ Wt,         // [64,64] n-major, f16
    float* __restrict__ cinit)         // [64]
{
  __shared__ float rs[4][64], rq[4][64];
  __shared__ float sscale[64], sshift[64];
  const int col = threadIdx.x & 63;
  const int rg  = threadIdx.x >> 6;
  if (X) {
    float s = 0.f, q = 0.f;
    for (int r = rg; r < rows; r += 4) {
      float v = X[r * NF + col];
      s += v; q += v * v;
    }
    rs[rg][col] = s; rq[rg][col] = q;
  }
  __syncthreads();
  if (rg == 0) {
    float s, q;
    if (X) {
      s = rs[0][col] + rs[1][col] + rs[2][col] + rs[3][col];
      q = rq[0][col] + rq[1][col] + rq[2][col] + rq[3][col];
    } else {
      s = accum[col]; q = accum[64 + col];
    }
    const float inv  = 1.0f / (float)rows;
    const float mean = s * inv;
    const float var  = q * inv - mean * mean;         // biased variance
    const float sc   = g[col] * rsqrtf(var + 1e-5f);
    sscale[col] = sc;
    sshift[col] = b[col] - mean * sc;
  }
  __syncthreads();
  for (int i = threadIdx.x; i < 64 * 64; i += 256) {
    const int n = i >> 6, k = i & 63;
    Wt[n * 64 + k] = (_Float16)(sscale[k] * W[k * 64 + n]);
  }
  if (threadIdx.x < 64) {
    const int n = threadIdx.x;
    float acc = bias ? bias[n] : 0.0f;
    for (int k = 0; k < 64; ++k) acc += sshift[k] * W[k * 64 + n];
    cinit[n] = acc;
  }
}

// ----------------------------------------------------------- WMMA GEMM [rows,64]x[64,64]
// One 16-row x 64-col tile per wave32; K=64 => 2x v_wmma_f32_16x16x32_f16 per 16x16 tile.
// All A/B fragments preloaded up front (single load batch), then 8 back-to-back WMMAs.
template<int RELU, int ZAGG>
__global__ __launch_bounds__(256) void gemm64_wmma(
    const float* __restrict__ X, int rows,
    const _Float16* __restrict__ Wt,   // [64][64] n-major, BN folded
    const float* __restrict__ cinit,   // [64]
    float* __restrict__ OUT,
    float* __restrict__ AGG)           // zeroed when ZAGG
{
  const int wave = threadIdx.x >> 5;
  const int lane = threadIdx.x & 31;
  const int tile = blockIdx.x * 8 + wave;
  if (tile * 16 >= rows) return;
  const int l15  = lane & 15;
  const int hi   = lane >> 4;          // 0: K-low half, 1: K-high half
  const int row0 = tile * 16;
  const int arow = row0 + l15;         // A row this lane loads
  const int c0   = hi * 16;

  // ---- preload everything: A rows, all 8 B fragments, 4 cinit scalars ----
  const float* xr = X + (size_t)arow * NF;
  v4f u0 = *(const v4f*)(xr + c0 +  0), u1 = *(const v4f*)(xr + c0 +  4);
  v4f u2 = *(const v4f*)(xr + c0 +  8), u3 = *(const v4f*)(xr + c0 + 12);
  v4f u4 = *(const v4f*)(xr + c0 + 32), u5 = *(const v4f*)(xr + c0 + 36);
  v4f u6 = *(const v4f*)(xr + c0 + 40), u7 = *(const v4f*)(xr + c0 + 44);

  v16h bf0[4], bf1[4];
  float ci[4];
#pragma unroll
  for (int nt = 0; nt < 4; ++nt) {
    const int col = nt * 16 + l15;
    bf0[nt] = *(const v16h*)(Wt + col * NF + hi * 16);        // K chunk 0..31
    bf1[nt] = *(const v16h*)(Wt + col * NF + 32 + hi * 16);   // K chunk 32..63
    ci[nt]  = cinit[col];
  }

  // A fragments (16-bit A 16x32, contiguous k order):
  //  elems[0..15]=K[c0..c0+15] (c0 = 16*(lane/16)), matching B
  v16h a0, a1;
#pragma unroll
  for (int j = 0; j < 4; ++j) {
    a0[j]      = (_Float16)u0[j];  a0[4 + j]  = (_Float16)u1[j];
    a0[8 + j]  = (_Float16)u2[j];  a0[12 + j] = (_Float16)u3[j];
    a1[j]      = (_Float16)u4[j];  a1[4 + j]  = (_Float16)u5[j];
    a1[8 + j]  = (_Float16)u6[j];  a1[12 + j] = (_Float16)u7[j];
  }

  // ---- 8 WMMAs back-to-back ----
  v8f c[4];
#pragma unroll
  for (int nt = 0; nt < 4; ++nt) {
    const float s = ci[nt];
    c[nt] = (v8f){s, s, s, s, s, s, s, s};
  }
#pragma unroll
  for (int nt = 0; nt < 4; ++nt)
    c[nt] = __builtin_amdgcn_wmma_f32_16x16x32_f16(false, a0, false, bf0[nt],
                                                   (short)0, c[nt], false, false);
#pragma unroll
  for (int nt = 0; nt < 4; ++nt)
    c[nt] = __builtin_amdgcn_wmma_f32_16x16x32_f16(false, a1, false, bf1[nt],
                                                   (short)0, c[nt], false, false);

  // ---- epilogue: C layout: VGPR r -> row r (lanes 0-15) / r+8 (lanes 16-31) ----
#pragma unroll
  for (int nt = 0; nt < 4; ++nt) {
    const int col = nt * 16 + l15;
    float* op = OUT + (size_t)(row0 + hi * 8) * NF + col;
#pragma unroll
    for (int r = 0; r < 8; ++r) {
      float v = c[nt][r];
      if (RELU) v = fmaxf(v, 0.0f);
      op[r * NF] = v;
    }
    if (ZAGG) {
      float* ap = AGG + (size_t)(row0 + hi * 8) * NF + col;
#pragma unroll
      for (int r = 0; r < 8; ++r) ap[r * NF] = 0.0f;
    }
  }
}

// ----------------------------------------------------------- edge aggregation
// 16 lanes x float4 per edge; fp32 atomics land in L2 (whole graph is L2-resident).
__global__ __launch_bounds__(256) void edge_agg(
    const int* __restrict__ src, const int* __restrict__ dst,
    const float* __restrict__ norm,
    const float* __restrict__ m, float* __restrict__ agg)
{
  const int gid = blockIdx.x * 256 + threadIdx.x;
  const int e = gid >> 4;
  if (e >= NEDGES) return;
  const int c = (gid & 15) << 2;
  const int s = src[e], d = dst[e];
  const float w = norm[e];
  v4f mv = *(const v4f*)(m + (size_t)s * NF + c);
  float* ap = agg + (size_t)d * NF + c;
  atomicAdd(ap + 0, w * mv[0]);
  atomicAdd(ap + 1, w * mv[1]);
  atomicAdd(ap + 2, w * mv[2]);
  atomicAdd(ap + 3, w * mv[3]);
}

// h += relu(agg + dinv^2 * m (self-loop) + conv_b)
__global__ __launch_bounds__(256) void resid_relu(
    float* __restrict__ h, const float* __restrict__ m,
    const float* __restrict__ agg, const float* __restrict__ dinv,
    const float* __restrict__ cb)
{
  const int t = blockIdx.x * 256 + threadIdx.x;
  if (t >= NNODES * 16) return;
  const int i = t >> 4, c = (t & 15) << 2;
  const float di = dinv[i];
  const float w = di * di;
  v4f hv = *(v4f*)(h + (size_t)i * NF + c);
  v4f mv = *(const v4f*)(m + (size_t)i * NF + c);
  v4f av = *(const v4f*)(agg + (size_t)i * NF + c);
  v4f bv = *(const v4f*)(cb + c);
#pragma unroll
  for (int j = 0; j < 4; ++j)
    hv[j] = hv[j] + fmaxf(av[j] + w * mv[j] + bv[j], 0.0f);
  *(v4f*)(h + (size_t)i * NF + c) = hv;
}

__global__ __launch_bounds__(256) void pool_add(
    const float* __restrict__ h, const int* __restrict__ batch,
    float* __restrict__ pooled)
{
  const int t = blockIdx.x * 256 + threadIdx.x;
  if (t >= NNODES * 16) return;
  const int i = t >> 4, c = (t & 15) << 2;
  const int gidx = batch[i];
  v4f hv = *(const v4f*)(h + (size_t)i * NF + c);
  float* pp = pooled + (size_t)gidx * NF + c;
  atomicAdd(pp + 0, hv[0]);
  atomicAdd(pp + 1, hv[1]);
  atomicAdd(pp + 2, hv[2]);
  atomicAdd(pp + 3, hv[3]);
}

__global__ __launch_bounds__(256) void final_bn(
    const float* __restrict__ X, const float* __restrict__ g,
    const float* __restrict__ b, float* __restrict__ out)
{
  __shared__ float rs[4][64], rq[4][64];
  __shared__ float sscale[64], sshift[64];
  const int col = threadIdx.x & 63;
  const int rg  = threadIdx.x >> 6;
  float s = 0.f, q = 0.f;
  for (int r = rg; r < NGRAPH; r += 4) {
    float v = X[r * NF + col];
    s += v; q += v * v;
  }
  rs[rg][col] = s; rq[rg][col] = q;
  __syncthreads();
  if (rg == 0) {
    s = rs[0][col] + rs[1][col] + rs[2][col] + rs[3][col];
    q = rq[0][col] + rq[1][col] + rq[2][col] + rq[3][col];
    const float mean = s / (float)NGRAPH;
    const float var  = q / (float)NGRAPH - mean * mean;
    const float sc   = g[col] * rsqrtf(var + 1e-5f);
    sscale[col] = sc;
    sshift[col] = b[col] - mean * sc;
  }
  __syncthreads();
  for (int i = threadIdx.x; i < NGRAPH * NF; i += 256) {
    const int c = i & 63;
    out[i] = X[i] * sscale[c] + sshift[c];
  }
}

// ----------------------------------------------------------------- launcher
extern "C" void kernel_launch(void* const* d_in, const int* in_sizes, int n_in,
                              void* d_out, int out_size, void* d_ws, size_t ws_size,
                              hipStream_t stream) {
  (void)in_sizes; (void)n_in; (void)out_size; (void)ws_size;
  const float* x     = (const float*)d_in[0];
  const int*   ei    = (const int*)d_in[1];
  const int*   batch = (const int*)d_in[2];
  const float* bnf_g = (const float*)d_in[3];
  const float* bnf_b = (const float*)d_in[4];
  const float* linW  = (const float*)d_in[5];
  const float* linb  = (const float*)d_in[6];
  const float* bnc_g = (const float*)d_in[7];
  const float* bnc_b = (const float*)d_in[8];
  const float* convW = (const float*)d_in[9];
  const float* convb = (const float*)d_in[10];
  const float* bnfc_g= (const float*)d_in[11];
  const float* bnfc_b= (const float*)d_in[12];
  const float* fcW   = (const float*)d_in[13];
  const float* fcb   = (const float*)d_in[14];
  const float* bnh_g = (const float*)d_in[15];
  const float* bnh_b = (const float*)d_in[16];
  const int* src = ei;
  const int* dst = ei + NEDGES;

  uint8_t* base = (uint8_t*)d_ws;
  size_t off = 0;
  auto carve = [&](size_t bytes) -> void* {
    void* p = base + off;
    off = (off + bytes + 255) & ~(size_t)255;
    return p;
  };
  float*    h     = (float*)carve((size_t)NNODES * NF * 4);
  float*    m     = (float*)carve((size_t)NNODES * NF * 4);
  float*    agg   = (float*)carve((size_t)NNODES * NF * 4);
  float*    norm  = (float*)carve((size_t)NEDGES * 4);
  float*    dinv  = (float*)carve((size_t)NNODES * 4);
  int*      degi  = (int*)carve((size_t)NNODES * 4);
  float*    accum = (float*)carve(128 * 4);
  _Float16* Wt    = (_Float16*)carve(64 * 64 * 2);
  float*    cin   = (float*)carve(64 * 4);
  float*    g0    = (float*)carve((size_t)NGRAPH * NF * 4);
  float*    g1    = (float*)carve((size_t)NGRAPH * NF * 4);

  const int bN16    = (NNODES + 255) / 256;
  const int bE      = (NEDGES + 255) / 256;
  const int bE16    = (NEDGES * 16 + 255) / 256;
  const int bNode16 = (NNODES * 16 + 255) / 256;
  const int gemmN   = (NNODES / 16 + 7) / 8;   // 8 waves/block, 1 tile/wave
  const int gemmG   = (NGRAPH / 16 + 7) / 8;

  // degrees + symmetric norm (with self-loops), norm precomputed once
  zero_f32<<<bN16, 256, 0, stream>>>((float*)degi, NNODES);
  deg_count<<<bE, 256, 0, stream>>>(dst, degi);
  deg_inv<<<bN16, 256, 0, stream>>>(degi, dinv);
  edge_norm<<<bE, 256, 0, stream>>>(src, dst, dinv, norm);

  // stem: h = relu(BN(x) @ linW + linb)
  zero_f32<<<1, 256, 0, stream>>>(accum, 128);
  stats_accum<<<512, 256, 0, stream>>>(x, NNODES, accum);
  fold_params<<<1, 256, 0, stream>>>(accum, nullptr, NNODES, bnf_g, bnf_b, linW, linb, Wt, cin);
  gemm64_wmma<1, 0><<<gemmN, 256, 0, stream>>>(x, NNODES, Wt, cin, h, nullptr);

  // conv layers
  for (int l = 0; l < NLAYERS; ++l) {
    zero_f32<<<1, 256, 0, stream>>>(accum, 128);
    stats_accum<<<512, 256, 0, stream>>>(h, NNODES, accum);
    fold_params<<<1, 256, 0, stream>>>(accum, nullptr, NNODES,
                                       bnc_g + 64 * l, bnc_b + 64 * l,
                                       convW + 4096 * l, nullptr, Wt, cin);
    gemm64_wmma<0, 1><<<gemmN, 256, 0, stream>>>(h, NNODES, Wt, cin, m, agg);
    edge_agg<<<bE16, 256, 0, stream>>>(src, dst, norm, m, agg);
    resid_relu<<<bNode16, 256, 0, stream>>>(h, m, agg, dinv, convb + 64 * l);
  }

  // global add pool
  zero_f32<<<(NGRAPH * NF + 255) / 256, 256, 0, stream>>>(g0, NGRAPH * NF);
  pool_add<<<bNode16, 256, 0, stream>>>(h, batch, g0);

  // fc head
  fold_params<<<1, 256, 0, stream>>>(nullptr, g0, NGRAPH, bnfc_g, bnfc_b, fcW, fcb, Wt, cin);
  gemm64_wmma<1, 0><<<gemmG, 256, 0, stream>>>(g0, NGRAPH, Wt, cin, g1, nullptr);
  fold_params<<<1, 256, 0, stream>>>(nullptr, g1, NGRAPH, bnfc_g + 64, bnfc_b + 64,
                                     fcW + 4096, fcb + 64, Wt, cin);
  gemm64_wmma<1, 0><<<gemmG, 256, 0, stream>>>(g1, NGRAPH, Wt, cin, g0, nullptr);

  // final BN -> out [512,64]
  final_bn<<<1, 256, 0, stream>>>(g0, bnh_g, bnh_b, (float*)d_out);
}
